// CrossAttention_7980049236060
// MI455X (gfx1250) — compile-verified, matrix operand order checked by reading, not run on hardware
//
#include <hip/hip_runtime.h>

// ---------- fixed problem sizes ----------
constexpr int kB   = 4;
constexpr int kT   = 4096;
constexpr int kS   = 1024;
constexpr int kHID = 1024;
constexpr int kCTX = 768;
constexpr int kEMB = 1024;
constexpr int kH   = 16;
constexpr int kD   = 64;   // head dim

// ---------- types ----------
typedef __attribute__((ext_vector_type(8)))  __bf16 bf16x8;
typedef __attribute__((ext_vector_type(16))) __bf16 v16bf;
typedef __attribute__((ext_vector_type(8)))  float  v8f;
typedef __attribute__((ext_vector_type(4)))  float  f32x4;

__device__ __forceinline__ __bf16 f2bf(float f) {
  unsigned u = __builtin_bit_cast(unsigned, f);
  u += 0x7fffu + ((u >> 16) & 1u);          // round-to-nearest-even
  unsigned short s = (unsigned short)(u >> 16);
  return __builtin_bit_cast(__bf16, s);
}

__device__ __forceinline__ v8f wmma_bf16(v16bf a, v16bf b, v8f c) {
  // D = A(16x32) * B(32x16) + C, f32 accumulate
  return __builtin_amdgcn_wmma_f32_16x16x32_bf16(
      false, a, false, b, (short)0, c, false, false);
}

// Load a 16-element bf16 operand fragment for one lane.
// p points at (row-or-col base + kk). ISA layout: element e holds
// k = e%8 + 8*(lane/16) + 16*(e/8)  -> two contiguous 16B loads that land
// directly in the operand's VGPR layout.
__device__ __forceinline__ v16bf load_frag(const __bf16* p, int half) {
  union { v16bf v; bf16x8 h[2]; } u;
  u.h[0] = *(const bf16x8*)(p + (half << 3));
  u.h[1] = *(const bf16x8*)(p + 16 + (half << 3));
  return u.v;
}

// ---------- conversion kernels ----------
// 8 elements per thread: two b128 loads -> one b128 store (pure HBM streaming)
__global__ void f32_to_bf16_kernel(const float* __restrict__ src,
                                   __bf16* __restrict__ dst, int n8) {
  int i = blockIdx.x * blockDim.x + threadIdx.x;
  if (i < n8) {
    f32x4 a = *(const f32x4*)(src + i * 8);
    f32x4 b = *(const f32x4*)(src + i * 8 + 4);
    bf16x8 o;
#pragma unroll
    for (int j = 0; j < 4; ++j) { o[j] = f2bf(a[j]); o[j + 4] = f2bf(b[j]); }
    *(bf16x8*)(dst + i * 8) = o;
  }
}

// src [K x N] row-major f32 -> dst [N x K] row-major bf16
__global__ void transpose_f32_to_bf16_kernel(const float* __restrict__ src,
                                             __bf16* __restrict__ dst,
                                             int K, int N) {
  int i = blockIdx.x * blockDim.x + threadIdx.x;
  if (i < K * N) {
    int k = i / N, n = i - k * N;
    dst[(size_t)n * K + k] = f2bf(src[i]);
  }
}

// ---------- WMMA GEMM: one wave computes a 32(M) x 64(N) tile ----------
// A  : bf16 [M x K] row-major
// Bt : bf16 [N x K] row-major (B transposed -> contiguous frag loads)
// MODE 0: f32 out [M x N] += bias[n]
// MODE 1: bf16 out [M x N]
// MODE 2: bf16 out as V-transposed [B,H,D,S]  (M = kB*kS rows, N = kEMB cols)
template <int MODE>
__global__ __launch_bounds__(32) void gemm32x64_kernel(
    const __bf16* __restrict__ A, const __bf16* __restrict__ Bt,
    const float* __restrict__ bias, void* __restrict__ Cout, int K, int N) {
  const int lane = threadIdx.x;
  const int lane16 = lane & 15;
  const int half = lane >> 4;

  const __bf16* arow[2];
#pragma unroll
  for (int mt = 0; mt < 2; ++mt)
    arow[mt] = A + (size_t)(blockIdx.x * 32 + mt * 16 + lane16) * K;
  const __bf16* brow[4];
#pragma unroll
  for (int t = 0; t < 4; ++t)
    brow[t] = Bt + (size_t)(blockIdx.y * 64 + t * 16 + lane16) * K;

  v8f acc[2][4] = {};
  for (int kk = 0; kk < K; kk += 32) {
    v16bf a0 = load_frag(arow[0] + kk, half);
    v16bf a1 = load_frag(arow[1] + kk, half);
#pragma unroll
    for (int t = 0; t < 4; ++t) {
      v16bf b = load_frag(brow[t] + kk, half);
      acc[0][t] = wmma_bf16(a0, b, acc[0][t]);
      acc[1][t] = wmma_bf16(a1, b, acc[1][t]);
    }
  }

#pragma unroll
  for (int mt = 0; mt < 2; ++mt) {
#pragma unroll
    for (int t = 0; t < 4; ++t) {
#pragma unroll
      for (int r = 0; r < 8; ++r) {
        int m = blockIdx.x * 32 + mt * 16 + r + half * 8;  // row = r + 8*half
        int n = blockIdx.y * 64 + t * 16 + lane16;         // col = lane%16
        float v = acc[mt][t][r];
        if (MODE == 0) {
          ((float*)Cout)[(size_t)m * N + n] = v + bias[n];
        } else if (MODE == 1) {
          ((__bf16*)Cout)[(size_t)m * N + n] = f2bf(v);
        } else {
          int b = m >> 10, s = m & (kS - 1);               // kS = 1024
          int h = n >> 6,  d = n & (kD - 1);               // kD = 64
          ((__bf16*)Cout)[((size_t)((b * kH + h) * kD + d) << 10) + s] = f2bf(v);
        }
      }
    }
  }
}

// ---------- flash attention ----------
// Workgroup = 128 threads (4 waves) handles (b, h, 64 query rows).
// Each 32-key chunk: the workgroup stages K (32x64) and V (64x32) tiles into
// LDS once, then all 4 waves (16 query rows each) build their WMMA B-operands
// from LDS. K/V global traffic is amortized over 64 query rows; per-wave
// register pressure stays well under 256 VGPRs.
// Q  : bf16 [B*T, EMB]   (row t, col h*64+d)
// Km : bf16 [B*S, EMB]
// Vt : bf16 [B, H, D, S]
// Ctx: bf16 [B*T, EMB]
__global__ __launch_bounds__(128) void attn_fwd_kernel(
    const __bf16* __restrict__ Q, const __bf16* __restrict__ Km,
    const __bf16* __restrict__ Vt, __bf16* __restrict__ Ctx) {
  __shared__ __bf16 Kl[32 * 64];      // [s][d] : score B-operand, ld = 64
  __shared__ __bf16 Vl[64 * 32];      // [d][s] : PV B-operand,    ld = 32
  __shared__ __bf16 Pl[4][16 * 32];   // per-wave P tile, [row][k], ld = 32

  const int tid = threadIdx.x;
  const int wv = tid >> 5;            // wave id 0..3
  const int lane = tid & 31;
  const int lane16 = lane & 15;
  const int half = (lane >> 4) & 1;
  const int b = blockIdx.z, h = blockIdx.y;
  const int tw = blockIdx.x * 64 + wv * 16;   // this wave's first query row

  // Q operand fragments for full K-dim D=64 (two 32-wide k-steps)
  const __bf16* qrow = Q + (size_t)(b * kT + tw + lane16) * kEMB + h * kD;
  const v16bf aq0 = load_frag(qrow, half);
  const v16bf aq1 = load_frag(qrow + 32, half);

  const __bf16* Kbase = Km + (size_t)(b * kS) * kEMB + h * kD;
  const __bf16* Vbase = Vt + (size_t)((b * kH + h) * kD) * kS;

  float mi[8], li[8];
  v8f o[4] = {};
#pragma unroll
  for (int r = 0; r < 8; ++r) { mi[r] = -__builtin_inff(); li[r] = 0.f; }

  for (int s0 = 0; s0 < kS; s0 += 32) {
    // ---- cooperative staging: K tile 4KB + V tile 4KB, 16B units ----
#pragma unroll
    for (int u0 = 0; u0 < 2; ++u0) {
      int u = tid + u0 * 128;                     // 0..255
      int krow = u >> 3, kcol = (u & 7) * 8;      // 8 units per 64-elem K row
      *(bf16x8*)&Kl[krow * 64 + kcol] =
          *(const bf16x8*)(Kbase + (size_t)(s0 + krow) * kEMB + kcol);
      int vrow = u >> 2, vcol = (u & 3) * 8;      // 4 units per 32-elem V row
      *(bf16x8*)&Vl[vrow * 32 + vcol] =
          *(const bf16x8*)(Vbase + (size_t)vrow * kS + s0 + vcol);
    }
    __syncthreads();

    // ---- scores: 16x32 chunk = two 16x16 WMMA tiles, K-dim = D = 64 ----
    v8f sc[2] = {};
#pragma unroll
    for (int ch = 0; ch < 2; ++ch) {
      const __bf16* kb = &Kl[(ch * 16 + lane16) * 64];
      sc[ch] = wmma_bf16(aq0, load_frag(kb, half), sc[ch]);
      sc[ch] = wmma_bf16(aq1, load_frag(kb + 32, half), sc[ch]);
    }

    // ---- online softmax over the 32-key chunk ----
#pragma unroll
    for (int r = 0; r < 8; ++r) {
      float x0 = sc[0][r] * 0.125f;               // 1/sqrt(64)
      float x1 = sc[1][r] * 0.125f;
      float cm = fmaxf(x0, x1);
#pragma unroll
      for (int msk = 8; msk >= 1; msk >>= 1)
        cm = fmaxf(cm, __shfl_xor(cm, msk, 16));  // row reduce in half-group
      float mnew = fmaxf(mi[r], cm);
      float alpha = __expf(mi[r] - mnew);         // first chunk: exp(-inf)=0
      float p0 = __expf(x0 - mnew);
      float p1 = __expf(x1 - mnew);
      float rs = p0 + p1;
#pragma unroll
      for (int msk = 8; msk >= 1; msk >>= 1)
        rs += __shfl_xor(rs, msk, 16);
      li[r] = li[r] * alpha + rs;
      mi[r] = mnew;
#pragma unroll
      for (int t = 0; t < 4; ++t) o[t][r] *= alpha;
      // stage P into this wave's LDS tile, [row][k] layout for the A operand
      Pl[wv][(r + half * 8) * 32 + lane16]      = f2bf(p0);
      Pl[wv][(r + half * 8) * 32 + 16 + lane16] = f2bf(p1);
    }

    // ---- O += P(16x32) @ V(32x64): 4 WMMAs, K-dim = 32 exactly ----
    v16bf ap = load_frag(&Pl[wv][lane16 * 32], half);
#pragma unroll
    for (int t = 0; t < 4; ++t) {
      v16bf bv = load_frag(&Vl[(t * 16 + lane16) * 32], half);
      o[t] = wmma_bf16(ap, bv, o[t]);
    }
    __syncthreads();   // Kl/Vl consumed; safe to restage next chunk
  }

  // ---- normalize and store ctx (bf16) ----
#pragma unroll
  for (int t = 0; t < 4; ++t) {
#pragma unroll
    for (int r = 0; r < 8; ++r) {
      int row = tw + r + half * 8;
      int col = h * kD + t * 16 + lane16;
      Ctx[(size_t)(b * kT + row) * kEMB + col] = f2bf(o[t][r] / li[r]);
    }
  }
}

// ---------- host driver ----------
extern "C" void kernel_launch(void* const* d_in, const int* in_sizes, int n_in,
                              void* d_out, int out_size, void* d_ws, size_t ws_size,
                              hipStream_t stream) {
  (void)in_sizes; (void)n_in; (void)out_size; (void)ws_size;
  const float* tokens  = (const float*)d_in[0];  // [B,T,HID]
  const float* context = (const float*)d_in[1];  // [B,S,CTX]
  const float* Wq      = (const float*)d_in[2];  // [HID,EMB]
  const float* Wk      = (const float*)d_in[3];  // [CTX,EMB]
  const float* Wv      = (const float*)d_in[4];  // [CTX,EMB]
  const float* Wo      = (const float*)d_in[5];  // [EMB,HID]
  const float* bo      = (const float*)d_in[6];  // [HID]
  float* out = (float*)d_out;

  char* ws = (char*)d_ws;
  size_t off = 0;
  auto take = [&](size_t bytes) -> char* {
    char* p = ws + off;
    off += (bytes + 255) & ~(size_t)255;
    return p;
  };

  const int MT = kB * kT;   // 16384 query rows
  const int MS = kB * kS;   // 4096 key/value rows

  __bf16* tokbf = (__bf16*)take((size_t)MT * kHID * 2);
  __bf16* ctxbf = (__bf16*)take((size_t)MS * kCTX * 2);
  __bf16* WqT   = (__bf16*)take((size_t)kEMB * kHID * 2);
  __bf16* WkT   = (__bf16*)take((size_t)kEMB * kCTX * 2);
  __bf16* WvT   = (__bf16*)take((size_t)kEMB * kCTX * 2);
  __bf16* WoT   = (__bf16*)take((size_t)kHID * kEMB * 2);
  __bf16* Qb    = (__bf16*)take((size_t)MT * kEMB * 2);
  __bf16* Kb    = (__bf16*)take((size_t)MS * kEMB * 2);
  __bf16* Vt    = (__bf16*)take((size_t)MS * kEMB * 2);
  __bf16* Ctxb  = tokbf;    // alias: tokens-bf16 dead after the Q projection

  // 1) fp32 -> bf16 conversions (8 elems/thread, b128 in / b128 out)
  {
    int n8 = (MT * kHID) / 8;
    f32_to_bf16_kernel<<<(n8 + 255) / 256, 256, 0, stream>>>(tokens, tokbf, n8);
  }
  {
    int n8 = (MS * kCTX) / 8;
    f32_to_bf16_kernel<<<(n8 + 255) / 256, 256, 0, stream>>>(context, ctxbf, n8);
  }
  transpose_f32_to_bf16_kernel<<<(kHID * kEMB + 255) / 256, 256, 0, stream>>>(
      Wq, WqT, kHID, kEMB);
  transpose_f32_to_bf16_kernel<<<(kCTX * kEMB + 255) / 256, 256, 0, stream>>>(
      Wk, WkT, kCTX, kEMB);
  transpose_f32_to_bf16_kernel<<<(kCTX * kEMB + 255) / 256, 256, 0, stream>>>(
      Wv, WvT, kCTX, kEMB);
  transpose_f32_to_bf16_kernel<<<(kEMB * kHID + 255) / 256, 256, 0, stream>>>(
      Wo, WoT, kEMB, kHID);

  // 2) projections (WMMA bf16, f32 accumulate)
  gemm32x64_kernel<1><<<dim3(MT / 32, kEMB / 64), 32, 0, stream>>>(
      tokbf, WqT, nullptr, Qb, kHID, kEMB);
  gemm32x64_kernel<1><<<dim3(MS / 32, kEMB / 64), 32, 0, stream>>>(
      ctxbf, WkT, nullptr, Kb, kCTX, kEMB);
  gemm32x64_kernel<2><<<dim3(MS / 32, kEMB / 64), 32, 0, stream>>>(
      ctxbf, WvT, nullptr, Vt, kCTX, kEMB);

  // 3) flash attention: 4 waves x 16 query rows share K/V tiles via LDS
  attn_fwd_kernel<<<dim3(kT / 64, kH, kB), 128, 0, stream>>>(Qb, Kb, Vt, Ctxb);

  // 4) output projection + bias (fp32 out)
  gemm32x64_kernel<0><<<dim3(MT / 32, kHID / 64), 32, 0, stream>>>(
      Ctxb, WoT, bo, out, kEMB, kHID);
}
